// TreeLstmDecoderComplete_47047071760735
// MI455X (gfx1250) — compile-verified, matrix operand order checked by reading, not run on hardware
//
#include <hip/hip_runtime.h>
#include <math.h>

// ---------------- problem constants (from the reference) ----------------
#define LATN 512
#define NPT  85
#define NB   128
#define NTOT (NB * NPT)          // 10880 nodes

typedef __bf16 bf16;
typedef __attribute__((ext_vector_type(16))) bf16  v16bf;
typedef __attribute__((ext_vector_type(8)))  bf16  v8bf;
typedef __attribute__((ext_vector_type(4)))  bf16  v4bf;
typedef __attribute__((ext_vector_type(8)))  float v8f;

__device__ __forceinline__ bf16 f2bf(float f) {
  union { float f; unsigned u; } v; v.f = f;
  unsigned r = (v.u + 0x7FFFu + ((v.u >> 16) & 1u)) >> 16;  // round-to-nearest-even
  union { unsigned short s; bf16 b; } o; o.s = (unsigned short)r;
  return o.b;
}

// K offset inside a 32-wide chunk for WMMA bf16 A/B fragments (ISA 7.12.2):
// lanes 0-15 hold K 0..7 (V0-3) and 16..23 (V4-7); lanes 16-31 hold 8..15 / 24..31.
// => one fragment == two contiguous 16B runs: [base+half, +8) and [base+16+half, +8).
__device__ __forceinline__ v16bf frag_ld(const bf16* p, int lane) {
  const bf16* q = p + ((lane & 16) >> 1);      // +0 or +8
  v8bf c1 = *(const v8bf*)(q);
  v8bf c2 = *(const v8bf*)(q + 16);
  v16bf f;
#pragma unroll
  for (int i = 0; i < 8; ++i) { f[i] = c1[i]; f[8 + i] = c2[i]; }
  return f;
}

__device__ __forceinline__ v8f wmma_bf16(v16bf a, v16bf b, v8f c) {
  return __builtin_amdgcn_wmma_f32_16x16x32_bf16(false, a, false, b, (short)0, c,
                                                 false, false);
}

// gfx1250 async global->LDS copy (16B per lane), tracked by ASYNCcnt.
__device__ __forceinline__ void async_b128(const bf16* gsrc, bf16* ldst) {
  unsigned lds = (unsigned)(unsigned long long)(void*)ldst;  // low 32 bits = LDS offset
  unsigned long long ga = (unsigned long long)(const void*)gsrc;
  asm volatile("global_load_async_to_lds_b128 %0, %1, off" :: "v"(lds), "v"(ga)
               : "memory");
}
__device__ __forceinline__ void async_wait0() {
  asm volatile("s_wait_asynccnt 0" ::: "memory");
}

__device__ __forceinline__ float sigmf(float x) { return 1.f / (1.f + __expf(-x)); }

// ---------------- output accumulator init ----------------
__global__ void zero_out(float* o) {
  if (threadIdx.x == 0 && blockIdx.x == 0) o[0] = 0.f;
}

// ---------------- one-time f32 -> bf16 weight conversion ----------------
__global__ __launch_bounds__(256) void cvt_bf16(const float* __restrict__ s,
                                                bf16* __restrict__ d, int n4) {
  int i = blockIdx.x * 256 + threadIdx.x;
  if (i < n4) {
    float4 f = ((const float4*)s)[i];
    v4bf o;
    o[0] = f2bf(f.x); o[1] = f2bf(f.y); o[2] = f2bf(f.z); o[3] = f2bf(f.w);
    ((v4bf*)d)[i] = o;
  }
}

// ---------------- fused gather-GEMM: C = epi(A_gathered @ [W0|W1]^T) ----------------
struct GemmArgs {
  const float* z;  const float* h_p; const float* h_s;
  const float* emb; const int* feat;
  const bf16* W0; const bf16* W1;     // each (Nc, 512) row-major, bf16
  const float* b0; const float* b1;   // biases (Nc,) or null
  float* C;
  int M, Nc;
  int curbase, parbase, use_z, has_prev;
  int a0_emb;     // half0: 1 = embedding gather, 0 = parent state
  int a1_parent;  // half1: 1 = parent state,     0 = prev-sibling state
  int epi_tanh;   // 1 = tanh (h_pred), 0 = +bias (LSTM gates)
};

template <int NG>
__global__ __launch_bounds__(256) void gemm_gather(GemmArgs a) {
  __shared__ __align__(16) bf16 lA[64 * 40];    // 64 rows x 32 K (pad 40)
  __shared__ __align__(16) bf16 lB[128 * 40];   // 128 cols x 32 K
  const int tid  = threadIdx.x;
  const int lane = tid & 31, w = tid >> 5;
  const int wm = w >> 2, wn = w & 3;            // 2x4 wave grid -> 32x32 per wave
  const int bM = blockIdx.x * 64, bN = blockIdx.y * 128;
  v8f acc[2][2] = {};

  for (int k0 = 0; k0 < 1024; k0 += 32) {
    // --- stage B tile asynchronously (raw bf16 copy, 16B/lane chunks) ---
    {
      const bf16* Wsel = (k0 < LATN) ? a.W0 : a.W1;
      int kbase = (k0 < LATN) ? k0 : k0 - LATN;
#pragma unroll
      for (int e = tid; e < 512; e += 256) {     // 128 cols x 4 chunks
        int c = e >> 2, q = e & 3;
        async_b128(Wsel + (size_t)(bN + c) * LATN + kbase + q * 8,
                   &lB[c * 40 + q * 8]);
      }
    }
    // --- stage A tile (gathered, f32 -> bf16) ---
    for (int e = tid; e < 64 * 32; e += 256) {
      int r = e >> 5, kk = e & 31;
      int row = bM + r;
      int bb = row / NG, gg = row % NG;          // NG in {1,4,16}: shifts
      int cur = bb * NPT + a.curbase + gg * 4;
      int kg = k0 + kk;
      float v;
      if (kg < LATN) {
        if (a.a0_emb) {
          int lab = a.feat[cur];
          v = a.emb[(size_t)lab * LATN + kg];
        } else if (a.use_z) {
          v = a.z[bb * 1024 + kg];
        } else {
          int par = bb * NPT + a.parbase + gg;
          v = a.h_p[(size_t)par * LATN + kg];
        }
      } else {
        int k2 = kg - LATN;
        if (a.a1_parent) {
          if (a.use_z) v = a.z[bb * 1024 + k2];
          else {
            int par = bb * NPT + a.parbase + gg;
            v = a.h_p[(size_t)par * LATN + k2];
          }
        } else {
          v = a.has_prev ? a.h_s[(size_t)(cur - 1) * LATN + k2] : 0.f;
        }
      }
      lA[r * 40 + kk] = f2bf(v);
    }
    async_wait0();
    __syncthreads();

    v16bf a0 = frag_ld(&lA[(wm * 32 +      (lane & 15)) * 40], lane);
    v16bf a1 = frag_ld(&lA[(wm * 32 + 16 + (lane & 15)) * 40], lane);
    v16bf b0 = frag_ld(&lB[(wn * 32 +      (lane & 15)) * 40], lane);
    v16bf b1 = frag_ld(&lB[(wn * 32 + 16 + (lane & 15)) * 40], lane);
    acc[0][0] = wmma_bf16(a0, b0, acc[0][0]);
    acc[0][1] = wmma_bf16(a0, b1, acc[0][1]);
    acc[1][0] = wmma_bf16(a1, b0, acc[1][0]);
    acc[1][1] = wmma_bf16(a1, b1, acc[1][1]);
    __syncthreads();
  }

  const int mh = (lane & 16) ? 8 : 0;   // C layout: lanes 16-31 hold M+8
#pragma unroll
  for (int mi = 0; mi < 2; ++mi)
#pragma unroll
    for (int nj = 0; nj < 2; ++nj)
#pragma unroll
      for (int r = 0; r < 8; ++r) {
        int row = bM + wm * 32 + mi * 16 + r + mh;
        int col = bN + wn * 32 + nj * 16 + (lane & 15);
        if (row < a.M) {
          float v = acc[mi][nj][r];
          if (a.epi_tanh) v = tanhf(v);
          else if (a.b0)  v += a.b0[col] + a.b1[col];
          a.C[(size_t)row * a.Nc + col] = v;
        }
      }
}

// ---------------- online-softmax cross-entropy over vocab (WMMA) ----------------
// 16 rows/block, vocab swept in 128-col chunks (8 waves x 16 cols).
// Per-row constant offsets (off_p*ip + off_s*hs) cancel in CE and are dropped.
template <int NG>
__global__ __launch_bounds__(256) void ce_rows(
    const float* __restrict__ hpred, const bf16* __restrict__ W,
    const float* __restrict__ bias, const int* __restrict__ feat, int vocab,
    float inv_norm, int curbase, float* __restrict__ out) {
  __shared__ __align__(16) bf16 lA[16 * 520];   // 16 rows x 512 K (pad)
  __shared__ float sm[8][16][3];                // per-wave (max, sumexp, label logit)
  const int tid = threadIdx.x;
  const int rowbase = blockIdx.x * 16;

  for (int e = tid; e < 16 * LATN; e += 256) {
    int r = e >> 9, k = e & 511;
    lA[r * 520 + k] = f2bf(hpred[(size_t)(rowbase + r) * LATN + k]);
  }
  __syncthreads();

  const int lane = tid & 31, wv = tid >> 5;
  const int l15 = lane & 15;
  const int mh = (lane & 16) ? 8 : 0;

  float m[8], ss[8], lab[8];
  int lbl[8];
#pragma unroll
  for (int r = 0; r < 8; ++r) {
    m[r] = -1e30f; ss[r] = 0.f; lab[r] = 0.f;
    int row = rowbase + r + mh;
    int bb = row / NG, gg = row % NG;
    lbl[r] = feat[bb * NPT + curbase + gg * 4];
  }

  const int nchunk = (vocab + 127) >> 7;
  for (int ch = 0; ch < nchunk; ++ch) {
    int v = (ch << 7) + (wv << 4) + l15;
    bool valid = v < vocab;
    int vc = valid ? v : 0;
    const bf16* wrow = W + (size_t)vc * LATN;
    if (ch + 1 < nchunk)                        // gfx1250 global_prefetch_b8
      __builtin_prefetch(wrow + 128 * LATN, 0, 1);
    v8f acc = {};
#pragma unroll 4
    for (int kk = 0; kk < 16; ++kk) {
      v16bf af = frag_ld(&lA[l15 * 520 + kk * 32], lane);
      v16bf bfr = frag_ld(wrow + kk * 32, lane);  // two b128 global loads
      acc = wmma_bf16(af, bfr, acc);
    }
    float bv = bias[vc];
#pragma unroll
    for (int r = 0; r < 8; ++r) {
      float x = acc[r] + bv;
      if (valid) {
        float mn = fmaxf(m[r], x);
        ss[r] = ss[r] * __expf(m[r] - mn) + __expf(x - mn);
        m[r] = mn;
        if (v == lbl[r]) lab[r] = x;
      }
    }
  }

  // merge the 16 lanes of each half (same rows, different columns)
#pragma unroll
  for (int r = 0; r < 8; ++r) {
#pragma unroll
    for (int msk = 1; msk < 16; msk <<= 1) {
      float om = __shfl_xor(m[r], msk, 32);
      float os = __shfl_xor(ss[r], msk, 32);
      float ol = __shfl_xor(lab[r], msk, 32);
      float mn = fmaxf(m[r], om);
      ss[r] = ss[r] * __expf(m[r] - mn) + os * __expf(om - mn);
      m[r] = mn;
      lab[r] += ol;
    }
    if (l15 == 0) {
      sm[wv][r + mh][0] = m[r];
      sm[wv][r + mh][1] = ss[r];
      sm[wv][r + mh][2] = lab[r];
    }
  }
  __syncthreads();
  if (tid < 16) {
    float M2 = -1e30f, S = 0.f, L = 0.f;
    for (int w2 = 0; w2 < 8; ++w2) {
      float om = sm[w2][tid][0], os = sm[w2][tid][1];
      float mn = fmaxf(M2, om);
      S = S * __expf(M2 - mn) + os * __expf(om - mn);
      M2 = mn;
      L += sm[w2][tid][2];
    }
    atomicAdd(out, (M2 + __logf(S) - L) * inv_norm);  // logsumexp - label logit
  }
}

// ---------------- BCE (depth & width heads), one wave per row ----------------
__global__ __launch_bounds__(256) void bce_rows(
    const float* __restrict__ hpred, const float* __restrict__ Wd,
    const float* __restrict__ bd, const float* __restrict__ Ww,
    const float* __restrict__ bw, float ip, float hsv, int n, float scale,
    float* __restrict__ out) {
  int lane = threadIdx.x & 31;
  int row = blockIdx.x * 8 + (threadIdx.x >> 5);
  if (row >= n) return;
  float dd = 0.f, dw = 0.f;
  const float* hr = hpred + (size_t)row * LATN;
#pragma unroll
  for (int k = lane; k < LATN; k += 32) {
    float h = hr[k];
    dd += h * Wd[k];
    dw += h * Ww[k];
  }
#pragma unroll
  for (int msk = 16; msk >= 1; msk >>= 1) {
    dd += __shfl_xor(dd, msk, 32);
    dw += __shfl_xor(dw, msk, 32);
  }
  if (lane == 0) {
    float xd = dd + bd[0], xw = dw + bw[0];
    float ld = fmaxf(xd, 0.f) - xd * ip  + log1pf(__expf(-fabsf(xd)));
    float lw = fmaxf(xw, 0.f) - xw * hsv + log1pf(__expf(-fabsf(xw)));
    atomicAdd(out, (ld + lw) * scale);
  }
}

// ---------------- LSTM gates -> (h, c), scattered into node state ----------------
template <int NG>
__global__ __launch_bounds__(256) void lstm_pointwise(
    const float* __restrict__ g, const float* __restrict__ z,
    const float* __restrict__ ctab, float* __restrict__ hout,
    float* __restrict__ cout, int curbase, int parbase, int use_z,
    int cold_parent, int has_prev, int n) {
  int t = blockIdx.x * 256 + threadIdx.x;
  int row = t >> 9, j = t & 511;
  if (row >= n) return;
  int bb = row / NG, gg = row % NG;
  int cur = bb * NPT + curbase + gg * 4;
  const float* gr = g + (size_t)row * 2048;
  float gi = gr[j], gf = gr[512 + j], gc = gr[1024 + j], go = gr[1536 + j];
  float cold;
  if (cold_parent)
    cold = use_z ? z[bb * 1024 + 512 + j]
                 : ctab[(size_t)(bb * NPT + parbase + gg) * LATN + j];
  else
    cold = has_prev ? ctab[(size_t)(cur - 1) * LATN + j] : 0.f;
  float c2 = sigmf(gf) * cold + sigmf(gi) * tanhf(gc);
  hout[(size_t)cur * LATN + j] = sigmf(go) * tanhf(c2);
  cout[(size_t)cur * LATN + j] = c2;
}

// ---------------- host: 13-step static schedule ----------------
extern "C" void kernel_launch(void* const* d_in, const int* in_sizes, int n_in,
                              void* d_out, int out_size, void* d_ws, size_t ws_size,
                              hipStream_t stream) {
  const float* z         = (const float*)d_in[0];
  const int*   features  = (const int*)  d_in[1];
  const float* U_parent  = (const float*)d_in[2];
  const float* U_sibling = (const float*)d_in[3];
  const float* W_depth   = (const float*)d_in[4];
  const float* b_depth   = (const float*)d_in[5];
  const float* W_width   = (const float*)d_in[6];
  const float* b_width   = (const float*)d_in[7];
  const float* Wih_p     = (const float*)d_in[8];
  const float* Whh_p     = (const float*)d_in[9];
  const float* bih_p     = (const float*)d_in[10];
  const float* bhh_p     = (const float*)d_in[11];
  const float* Wih_s     = (const float*)d_in[12];
  const float* Whh_s     = (const float*)d_in[13];
  const float* bih_s     = (const float*)d_in[14];
  const float* bhh_s     = (const float*)d_in[15];
  const float* emb_res   = (const float*)d_in[16];
  const float* emb_name  = (const float*)d_in[17];
  const float* predW_res = (const float*)d_in[18];
  const float* predb_res = (const float*)d_in[19];
  const float* predW_name= (const float*)d_in[20];
  const float* predb_name= (const float*)d_in[21];
  float* out = (float*)d_out;

  // workspace: node state + activations (f32, ~127 MB) + bf16 weights (~41 MB)
  float* ws = (float*)d_ws;
  const size_t S = (size_t)NTOT * LATN;
  float* h_p   = ws;
  float* c_p   = ws + S;
  float* h_s   = ws + 2 * S;
  float* c_s   = ws + 3 * S;
  float* hpred = ws + 4 * S;                       // 2048 x 512
  float* gP    = hpred + (size_t)2048 * 512;       // 2048 x 2048
  float* gS    = gP + (size_t)2048 * 2048;
  bf16*  wb    = (bf16*)(gS + (size_t)2048 * 2048);
  const size_t SQ = (size_t)512 * 512, SW = (size_t)2048 * 512;
  bf16* Upb  = wb;              bf16* Usb  = Upb + SQ;
  bf16* Wipb = Usb + SQ;        bf16* Whpb = Wipb + SW;
  bf16* Wisb = Whpb + SW;       bf16* Whsb = Wisb + SW;
  bf16* Wrb  = Whsb + SW;                          // 600  x 512
  bf16* Wnb  = Wrb + (size_t)600 * 512;            // 30000 x 512

  zero_out<<<dim3(1), dim3(32), 0, stream>>>(out);

  // one-shot weight conversion to bf16 (L2-resident afterwards)
  {
    struct Cv { const float* s; bf16* d; int n; };
    const Cv cv[8] = {
        {U_parent, Upb, 512 * 512},   {U_sibling, Usb, 512 * 512},
        {Wih_p, Wipb, 2048 * 512},    {Whh_p, Whpb, 2048 * 512},
        {Wih_s, Wisb, 2048 * 512},    {Whh_s, Whsb, 2048 * 512},
        {predW_res, Wrb, 600 * 512},  {predW_name, Wnb, 30000 * 512}};
    for (int i = 0; i < 8; ++i) {
      int n4 = cv[i].n / 4;
      cvt_bf16<<<dim3((n4 + 255) / 256), 256, 0, stream>>>(cv[i].s, cv[i].d, n4);
    }
  }

  struct StepH { int ng, curbase, parbase, use_z, has_prev, is_name; float ip, hs; };
  StepH steps[13];
  int ns = 0;
  steps[ns++] = {1, 0, 0, 1, 0, 0, 1.f, 0.f};      // root
  const int LOFF[4] = {0, 1, 5, 21};
  const int LSZ[4]  = {1, 4, 16, 64};
  for (int l = 1; l < 4; ++l)
    for (int s = 0; s < 4; ++s)
      steps[ns++] = {LSZ[l] / 4, LOFF[l] + s, LOFF[l - 1], 0, s > 0 ? 1 : 0,
                     l == 3 ? 1 : 0, l < 3 ? 1.f : 0.f, s < 3 ? 1.f : 0.f};

  const float bce_scale  = 1.f / (float)NTOT;   // /N
  const float res_scale  = 1.f / 2688.f;        // /N_RES
  const float name_scale = 1.f / 8192.f;        // /N_NAME

  for (int si = 0; si < 13; ++si) {
    StepH st = steps[si];
    int n = NB * st.ng;

    auto launch_gemm = [&](const GemmArgs& a, dim3 grd) {
      switch (st.ng) {
        case 1:  gemm_gather<1><<<grd, 256, 0, stream>>>(a);  break;
        case 4:  gemm_gather<4><<<grd, 256, 0, stream>>>(a);  break;
        default: gemm_gather<16><<<grd, 256, 0, stream>>>(a); break;
      }
    };
    auto launch_ce = [&](const bf16* W, const float* b, int vocab, float sc) {
      dim3 grd(n / 16);
      switch (st.ng) {
        case 1:  ce_rows<1><<<grd, 256, 0, stream>>>(hpred, W, b, features, vocab,
                                                     sc, st.curbase, out); break;
        case 4:  ce_rows<4><<<grd, 256, 0, stream>>>(hpred, W, b, features, vocab,
                                                     sc, st.curbase, out); break;
        default: ce_rows<16><<<grd, 256, 0, stream>>>(hpred, W, b, features, vocab,
                                                      sc, st.curbase, out); break;
      }
    };
    auto launch_pw = [&](const float* g, const float* ctab, float* ho, float* co,
                         int cold_parent) {
      dim3 grd(n * 2);
      switch (st.ng) {
        case 1:  lstm_pointwise<1><<<grd, 256, 0, stream>>>(g, z, ctab, ho, co,
                     st.curbase, st.parbase, st.use_z, cold_parent, st.has_prev, n);
                 break;
        case 4:  lstm_pointwise<4><<<grd, 256, 0, stream>>>(g, z, ctab, ho, co,
                     st.curbase, st.parbase, st.use_z, cold_parent, st.has_prev, n);
                 break;
        default: lstm_pointwise<16><<<grd, 256, 0, stream>>>(g, z, ctab, ho, co,
                     st.curbase, st.parbase, st.use_z, cold_parent, st.has_prev, n);
                 break;
      }
    };

    {   // h_pred = tanh([h_par | h_prev] @ [U_p | U_s]^T)
      GemmArgs a = {};
      a.z = z; a.h_p = h_p; a.h_s = h_s; a.emb = nullptr; a.feat = features;
      a.W0 = Upb; a.W1 = Usb; a.b0 = nullptr; a.b1 = nullptr;
      a.C = hpred; a.M = n; a.Nc = 512;
      a.curbase = st.curbase; a.parbase = st.parbase;
      a.use_z = st.use_z; a.has_prev = st.has_prev;
      a.a0_emb = 0; a.a1_parent = 0; a.epi_tanh = 1;
      launch_gemm(a, dim3(n / 64, 4));
    }

    bce_rows<<<dim3(n / 8), 256, 0, stream>>>(hpred, W_depth, b_depth, W_width,
                                              b_width, st.ip, st.hs, n, bce_scale, out);

    if (st.is_name) launch_ce(Wnb, predb_name, 30000, name_scale);
    else            launch_ce(Wrb, predb_res,  600,   res_scale);

    if (!st.is_name) {   // parent LSTM (internal levels only)
      GemmArgs a = {};
      a.z = z; a.h_p = h_p; a.h_s = h_s; a.emb = emb_res; a.feat = features;
      a.W0 = Wipb; a.W1 = Whpb; a.b0 = bih_p; a.b1 = bhh_p;
      a.C = gP; a.M = n; a.Nc = 2048;
      a.curbase = st.curbase; a.parbase = st.parbase;
      a.use_z = st.use_z; a.has_prev = st.has_prev;
      a.a0_emb = 1; a.a1_parent = 1; a.epi_tanh = 0;
      launch_gemm(a, dim3(n / 64, 16));
      launch_pw(gP, c_p, h_p, c_p, /*cold_parent=*/1);
    }

    {   // sibling LSTM (all levels)
      GemmArgs a = {};
      a.z = z; a.h_p = h_p; a.h_s = h_s;
      a.emb = st.is_name ? emb_name : emb_res; a.feat = features;
      a.W0 = Wisb; a.W1 = Whsb; a.b0 = bih_s; a.b1 = bhh_s;
      a.C = gS; a.M = n; a.Nc = 2048;
      a.curbase = st.curbase; a.parbase = st.parbase;
      a.use_z = st.use_z; a.has_prev = st.has_prev;
      a.a0_emb = 1; a.a1_parent = 0; a.epi_tanh = 0;
      launch_gemm(a, dim3(n / 64, 16));
      launch_pw(gS, c_s, h_s, c_s, /*cold_parent=*/0);
    }
  }
  (void)in_sizes; (void)n_in; (void)out_size; (void)ws_size;
}
